// GliomaGNN_18279380811878
// MI455X (gfx1250) — matrix operand order repaired and checked by reading, not compiled, not run on hardware
//
#include <hip/hip_runtime.h>

typedef __attribute__((ext_vector_type(16))) _Float16 v16h;
typedef __attribute__((ext_vector_type(8)))  float    v8f;

__device__ __forceinline__ float leaky02(float x) { return x > 0.f ? x : 0.2f * x; }
// monotone ordered key for float atomicMax via unsigned
__device__ __forceinline__ unsigned fkey(float x) {
  unsigned u = __float_as_uint(x);
  return (u >> 31) ? ~u : (u | 0x80000000u);
}
__device__ __forceinline__ float fdec(unsigned k) {
  return __uint_as_float((k >> 31) ? (k ^ 0x80000000u) : ~k);
}

// ---------------------------------------------------------------- zero fill
__global__ void k_zero(float* __restrict__ p, long long n) {
  long long i = (long long)blockIdx.x * blockDim.x + threadIdx.x;
  if (i < n) p[i] = 0.f;
}

// ------------------------------------------------- WMMA GEMM (f16 in, f32 acc)
// C[M,N] = act(A[M,K] @ W[K,N] + bias?) ; one wave computes a 16x16 tile.
// Assumes M % 16 == 0, N % 16 == 0, K % 32 == 0 (true for this model:
// M = 100000 = 6250*16, K in {32,64,128}, N in {32,128}).
// A fragment (16x32 f16, ISA 7.12.2): lane l: M=l%16, hf=l/16;
//   elem e<8 -> K = hf*8+e ; e>=8 -> K = 16 + hf*8 + (e-8)
//   => two contiguous 8-float runs per lane -> 4x b128 loads.
// B fragment (32x16 f16): lane l: N=l%16, hf=l/16; K order as A (e<8: hf*8+e ; e>=8: 16+hf*8+(e-8))
// C/D f32:  lane l: N=l%16; vgpr v -> M = v + 8*(l/16)
template <int KC>
__global__ void k_gemm_wmma(const float* __restrict__ A, const float* __restrict__ W,
                            const float* __restrict__ bias1, float* __restrict__ C,
                            int Mtiles, int N, int relu) {
  const int wv = threadIdx.x >> 5;              // wave id in block (4 waves)
  const int mt = blockIdx.x * 4 + wv;
  if (mt >= Mtiles) return;                     // whole-wave uniform exit
  const int m0   = mt * 16;
  const int n0   = blockIdx.y * 16;
  const int lane = threadIdx.x & 31;
  const int l16  = lane & 15;
  const int hf   = lane >> 4;
  const int colB = n0 + l16;

  // hoisted epilogue operands (single loads, no per-element branching)
  const float bv    = bias1 ? bias1[colB] : 0.f;
  const float floor = relu ? 0.f : -3.4e38f;

  const float* __restrict__ arow = A + (size_t)(m0 + l16) * KC;
  v8f acc = {0.f, 0.f, 0.f, 0.f, 0.f, 0.f, 0.f, 0.f};

#pragma unroll
  for (int k0 = 0; k0 < KC; k0 += 32) {
    const float4 a0 = *(const float4*)(arow + k0 + hf * 8);
    const float4 a1 = *(const float4*)(arow + k0 + hf * 8 + 4);
    const float4 a2 = *(const float4*)(arow + k0 + 16 + hf * 8);
    const float4 a3 = *(const float4*)(arow + k0 + 16 + hf * 8 + 4);
    v16h a, b;
    a[0]  = (_Float16)a0.x; a[1]  = (_Float16)a0.y; a[2]  = (_Float16)a0.z; a[3]  = (_Float16)a0.w;
    a[4]  = (_Float16)a1.x; a[5]  = (_Float16)a1.y; a[6]  = (_Float16)a1.z; a[7]  = (_Float16)a1.w;
    a[8]  = (_Float16)a2.x; a[9]  = (_Float16)a2.y; a[10] = (_Float16)a2.z; a[11] = (_Float16)a2.w;
    a[12] = (_Float16)a3.x; a[13] = (_Float16)a3.y; a[14] = (_Float16)a3.z; a[15] = (_Float16)a3.w;
    const float* __restrict__ wcol = W + (size_t)(k0 + hf * 8) * N + colB;
#pragma unroll
    for (int e = 0; e < 16; ++e) b[e] = (_Float16)wcol[(size_t)(e < 8 ? e : e + 8) * N];
    acc = __builtin_amdgcn_wmma_f32_16x16x32_f16(false, a, false, b,
                                                 (short)0, acc, false, false);
  }

  float* __restrict__ cp = C + (size_t)(m0 + 8 * hf) * N + colB;
#pragma unroll
  for (int v = 0; v < 8; ++v) {
    *cp = fmaxf(acc[v] + bv, floor);
    cp += N;
  }
}

// --------------------------------------------- attention score from hs buffer
// a_s[i,h] = sum_c hs[i,h,c] * att[h,c]
__global__ void k_att_src(const float* __restrict__ hs, const float* __restrict__ att,
                          float* __restrict__ as_, int Nn, int H, int C) {
  int t = blockIdx.x * blockDim.x + threadIdx.x;
  if (t >= Nn * H) return;
  int i = t / H, h = t - i * H;
  const float* row = hs + (size_t)i * H * C + h * C;
  float s = 0.f;
  for (int c = 0; c < C; ++c) s += row[c] * att[h * C + c];
  as_[t] = s;
}

// w_eff[k,h] = sum_c w_dst[k, h*C+c] * att_dst[h,c]   (tiny: din*H elements)
__global__ void k_weff(const float* __restrict__ wdst, const float* __restrict__ att,
                       float* __restrict__ weff, int din, int H, int C) {
  int t = blockIdx.x * blockDim.x + threadIdx.x;
  if (t >= din * H) return;
  int k = t / H, h = t - k * H;
  float s = 0.f;
  for (int c = 0; c < C; ++c) s += wdst[(size_t)k * H * C + h * C + c] * att[h * C + c];
  weff[t] = s;
}

// a_d[i,h] = x_dst[i,:] @ w_eff[:,h]   (avoids materializing hd)
__global__ void k_proj_att(const float* __restrict__ x, const float* __restrict__ weff,
                           float* __restrict__ ad, int Nn, int din, int H) {
  int t = blockIdx.x * blockDim.x + threadIdx.x;
  if (t >= Nn * H) return;
  int i = t / H, h = t - i * H;
  float s = 0.f;
  for (int k = 0; k < din; ++k) s += x[(size_t)i * din + k] * weff[k * H + h];
  ad[t] = s;
}

// ------------------------------------------------------- edge softmax passes
__global__ void k_edge_max(const int* __restrict__ ei, int E,
                           const float* __restrict__ as_, const float* __restrict__ ad_,
                           unsigned* __restrict__ mx, int H, int Ns, int Nd) {
  int t = blockIdx.x * blockDim.x + threadIdx.x;
  if (t >= E * H) return;
  int e = t / H, h = t - e * H;
  int s = max(0, min(ei[e], Ns - 1));
  int d = max(0, min(ei[E + e], Nd - 1));
  float logit = leaky02(as_[s * H + h] + ad_[d * H + h]);
  atomicMax(&mx[d * H + h], fkey(logit));
}

__global__ void k_edge_expsum(const int* __restrict__ ei, int E,
                              const float* __restrict__ as_, const float* __restrict__ ad_,
                              const unsigned* __restrict__ mx,
                              float* __restrict__ den, float* __restrict__ eexp,
                              int H, int Ns, int Nd) {
  int t = blockIdx.x * blockDim.x + threadIdx.x;
  if (t >= E * H) return;
  int e = t / H, h = t - e * H;
  int s = max(0, min(ei[e], Ns - 1));
  int d = max(0, min(ei[E + e], Nd - 1));
  float logit = leaky02(as_[s * H + h] + ad_[d * H + h]);
  float ex = __expf(logit - fdec(mx[d * H + h]));
  eexp[t] = ex;
  atomicAdd(&den[d * H + h], ex);
}

__global__ void k_edge_scatter(const int* __restrict__ ei, int E,
                               const float* __restrict__ hs, const float* __restrict__ eexp,
                               const float* __restrict__ den, float* __restrict__ acc,
                               int H, int C, int Ns, int Nd) {
  int t = blockIdx.x * blockDim.x + threadIdx.x;
  if (t >= E * H) return;
  int e = t / H, h = t - e * H;
  int s = max(0, min(ei[e], Ns - 1));
  int d = max(0, min(ei[E + e], Nd - 1));
  float alpha = eexp[t] / (den[d * H + h] + 1e-16f);
  const float* srcp = hs + (size_t)s * H * C + h * C;
  float* dstp = acc + (size_t)d * H * C + h * C;
  __builtin_prefetch(srcp, 0, 0);   // global_prefetch_b8
  for (int c = 0; c < C; ++c) atomicAdd(&dstp[c], srcp[c] * alpha);
}

// out = relu(acc + b1 (+ b2))
__global__ void k_finalize(const float* __restrict__ acc, const float* __restrict__ b1,
                           const float* __restrict__ b2, float* __restrict__ out,
                           long long n, int D) {
  long long i = (long long)blockIdx.x * blockDim.x + threadIdx.x;
  if (i >= n) return;
  int d = (int)(i % D);
  float v = acc[i] + b1[d] + (b2 ? b2[d] : 0.f);
  out[i] = fmaxf(v, 0.f);
}

// MLP head output: act( hid[i,:] @ w2 + b2 )
__global__ void k_head_out(const float* __restrict__ hid, const float* __restrict__ w2,
                           const float* __restrict__ b2, float* __restrict__ out,
                           int Nn, int act) {
  int i = blockIdx.x * blockDim.x + threadIdx.x;
  if (i >= Nn) return;
  float s = b2[0];
  for (int d = 0; d < 32; ++d) s += hid[(size_t)i * 32 + d] * w2[d];
  float r;
  if (act == 0)      r = 1.f / (1.f + __expf(-s));          // sigmoid
  else if (act == 1) r = fmaxf(s, 0.f);                      // relu
  else               r = (s > 20.f) ? s : log1pf(__expf(s)); // softplus
  out[i] = r;
}

struct GatP {
  const float *w_src, *w_dst, *att_src, *att_dst, *bias;
};

extern "C" void kernel_launch(void* const* d_in, const int* in_sizes, int n_in,
                              void* d_out, int out_size, void* d_ws, size_t ws_size,
                              hipStream_t stream) {
  if (n_in < 65) return;
  auto F = [&](int i) { return (const float*)d_in[i]; };

  const float* x_p = F(0);
  const float* x_t = F(1);
  const float* x_r = F(2);
  const int* ei_pt = (const int*)d_in[3];
  const int* ei_tp = (const int*)d_in[4];
  const int* ei_pr = (const int*)d_in[5];
  const int* ei_rp = (const int*)d_in[6];

  const long long NP = in_sizes[0] / 64;
  const long long NT = in_sizes[1] / 128;
  const long long NR = in_sizes[2] / 32;
  const int Ept = in_sizes[3] / 2, Etp = in_sizes[4] / 2;
  const int Epr = in_sizes[5] / 2, Erp = in_sizes[6] / 2;
  long long Nmax = NP > NT ? NP : NT; if (NR > Nmax) Nmax = NR;
  long long Emax = Ept; if (Etp > Emax) Emax = Etp; if (Epr > Emax) Emax = Epr; if (Erp > Emax) Emax = Erp;

  // ---- workspace layout (floats)
  float* w = (float*)d_ws;
  size_t o = 0;
  float* HP  = w + o; o += (size_t)NP * 32;
  float* HT  = w + o; o += (size_t)NT * 32;
  float* HR  = w + o; o += (size_t)NR * 32;
  float* H1P = w + o; o += (size_t)NP * 128;
  float* H1T = w + o; o += (size_t)NT * 128;
  float* H1R = w + o; o += (size_t)NR * 128;
  float* HS  = w + o; o += (size_t)Nmax * 128;   // per-GAT source projection
  float* ACC = w + o; o += (size_t)Nmax * 128;   // scatter accumulator
  float* AS  = w + o; o += (size_t)Nmax * 4;
  float* AD  = w + o; o += (size_t)Nmax * 4;
  float* DEN = w + o; o += (size_t)Nmax * 4;
  float* MAXK= w + o; o += (size_t)Nmax * 4;     // used as unsigned keys
  float* EEXP= w + o; o += (size_t)Emax * 4;
  float* WEFF= w + o; o += 128 * 4;
  if (ws_size < o * sizeof(float)) return;

  float* out = (float*)d_out;
  float* o_resp = out;
  float* o_surv = out + NR;
  float* o_unc  = out + 2 * NR;
  float* o_h2p  = out + 3 * NR;
  float* o_h2t  = o_h2p + NP * 32;
  float* o_h2r  = o_h2t + NT * 32;

  auto GB = [](long long n, int b) { return dim3((unsigned)((n + b - 1) / b)); };
  auto gemm = [&](const float* A, const float* W_, const float* b1, float* Cout,
                  int M, int K, int N, int relu) {
    const int Mtiles = M / 16;                      // exact for this model
    dim3 grid((Mtiles + 3) / 4, N / 16);
    dim3 blk(128);
    if (K == 32)
      k_gemm_wmma<32><<<grid, blk, 0, stream>>>(A, W_, b1, Cout, Mtiles, N, relu);
    else if (K == 64)
      k_gemm_wmma<64><<<grid, blk, 0, stream>>>(A, W_, b1, Cout, Mtiles, N, relu);
    else
      k_gemm_wmma<128><<<grid, blk, 0, stream>>>(A, W_, b1, Cout, Mtiles, N, relu);
  };
  auto gp = [&](int base) { return GatP{F(base), F(base + 1), F(base + 2), F(base + 3), F(base + 4)}; };

  auto run_gat = [&](const float* xsrc, const float* xdst, int Ns, int Nd, int din,
                     const int* ei, int Ecnt, const GatP& g, int H, int C) {
    const int D = H * C;
    gemm(xsrc, g.w_src, nullptr, HS, Ns, din, D, 0);                      // hs
    k_att_src<<<GB((long long)Ns * H, 256), 256, 0, stream>>>(HS, g.att_src, AS, Ns, H, C);
    k_weff<<<GB(din * H, 64), 64, 0, stream>>>(g.w_dst, g.att_dst, WEFF, din, H, C);
    k_proj_att<<<GB((long long)Nd * H, 256), 256, 0, stream>>>(xdst, WEFF, AD, Nd, din, H);
    k_zero<<<GB((long long)Nd * H, 256), 256, 0, stream>>>(MAXK, (long long)Nd * H);
    k_zero<<<GB((long long)Nd * H, 256), 256, 0, stream>>>(DEN, (long long)Nd * H);
    const long long EH = (long long)Ecnt * H;
    k_edge_max<<<GB(EH, 256), 256, 0, stream>>>(ei, Ecnt, AS, AD, (unsigned*)MAXK, H, Ns, Nd);
    k_edge_expsum<<<GB(EH, 256), 256, 0, stream>>>(ei, Ecnt, AS, AD, (const unsigned*)MAXK,
                                                   DEN, EEXP, H, Ns, Nd);
    k_edge_scatter<<<GB(EH, 256), 256, 0, stream>>>(ei, Ecnt, HS, EEXP, DEN, ACC, H, C, Ns, Nd);
  };
  auto zero = [&](float* p, long long n) { k_zero<<<GB(n, 256), 256, 0, stream>>>(p, n); };
  auto fin = [&](const float* b1, const float* b2, float* dst, long long Nn, int D) {
    k_finalize<<<GB(Nn * D, 256), 256, 0, stream>>>(ACC, b1, b2, dst, Nn * D, D);
  };

  // ---------------- encoders
  gemm(x_p, F(7),  F(8),  HP, (int)NP, 64,  32, 0);
  gemm(x_t, F(9),  F(10), HT, (int)NT, 128, 32, 0);
  gemm(x_r, F(11), F(12), HR, (int)NR, 32,  32, 0);

  GatP g1_pt = gp(13), g1_tp = gp(18), g1_pr = gp(23), g1_rp = gp(28);
  GatP g2_pt = gp(33), g2_tp = gp(38), g2_pr = gp(43), g2_rp = gp(48);

  // ---------------- layer 1 (H=4, C=32, din=32)
  zero(ACC, NT * 128);
  run_gat(HP, HT, (int)NP, (int)NT, 32, ei_pt, Ept, g1_pt, 4, 32);
  fin(g1_pt.bias, nullptr, H1T, NT, 128);

  zero(ACC, NP * 128);
  run_gat(HT, HP, (int)NT, (int)NP, 32, ei_tp, Etp, g1_tp, 4, 32);
  run_gat(HR, HP, (int)NR, (int)NP, 32, ei_rp, Erp, g1_rp, 4, 32);
  fin(g1_tp.bias, g1_rp.bias, H1P, NP, 128);

  zero(ACC, NR * 128);
  run_gat(HP, HR, (int)NP, (int)NR, 32, ei_pr, Epr, g1_pr, 4, 32);
  fin(g1_pr.bias, nullptr, H1R, NR, 128);

  // ---------------- layer 2 (H=1, C=32, din=128) -> straight into d_out
  zero(ACC, NT * 32);
  run_gat(H1P, H1T, (int)NP, (int)NT, 128, ei_pt, Ept, g2_pt, 1, 32);
  fin(g2_pt.bias, nullptr, o_h2t, NT, 32);

  zero(ACC, NP * 32);
  run_gat(H1T, H1P, (int)NT, (int)NP, 128, ei_tp, Etp, g2_tp, 1, 32);
  run_gat(H1R, H1P, (int)NR, (int)NP, 128, ei_rp, Erp, g2_rp, 1, 32);
  fin(g2_tp.bias, g2_rp.bias, o_h2p, NP, 32);

  zero(ACC, NR * 32);
  run_gat(H1P, H1R, (int)NP, (int)NR, 128, ei_pr, Epr, g2_pr, 1, 32);
  fin(g2_pr.bias, nullptr, o_h2r, NR, 32);

  // ---------------- heads on h2_r
  struct { int base; float* dst; int act; } heads[3] = {
      {53, o_resp, 0}, {57, o_surv, 1}, {61, o_unc, 2}};
  for (int i = 0; i < 3; ++i) {
    gemm(o_h2r, F(heads[i].base), F(heads[i].base + 1), HS, (int)NR, 32, 32, 1); // relu hidden
    k_head_out<<<GB(NR, 256), 256, 0, stream>>>(HS, F(heads[i].base + 2),
                                                F(heads[i].base + 3), heads[i].dst,
                                                (int)NR, heads[i].act);
  }
}
